// ClusteredSparsePattern_12120397709573
// MI455X (gfx1250) — matrix operand order repaired and checked by reading, not compiled, not run on hardware
//
#include <hip/hip_runtime.h>
#include <hip/hip_bf16.h>
#include <math.h>

typedef __attribute__((ext_vector_type(16))) _Float16 v16h;
typedef __attribute__((ext_vector_type(8)))  _Float16 v8h;
typedef __attribute__((ext_vector_type(8)))  float    v8f;
typedef __attribute__((ext_vector_type(4)))  float    v4f;

#define B_  2
#define H_  8
#define S_  2048
#define D_  64
#define C_  32
#define EPSN 1e-12f
#define THR  0.1f

// ---------------------------------------------------------------------------
// Kernel 1: L2-normalize rows, cosine-sim vs normalized centroids, softmax,
// emit f16 soft-assignments [B,H,S,C]. One thread per sequence row.
// ---------------------------------------------------------------------------
__global__ __launch_bounds__(256)
void assign_kernel(const float* __restrict__ x,      // [B,H,S,D]
                   const float* __restrict__ cents,  // [H,C,D]
                   const float* __restrict__ temp,   // [1]
                   _Float16*    __restrict__ assign) // [B,H,S,C] f16
{
    __shared__ float cent[C_ * D_];
    __shared__ float cinv[C_];

    const int bh = blockIdx.y;          // b*H + h
    const int h  = bh % H_;

    // Stage this head's centroids in LDS (8 KB), precompute inverse norms.
    const float* ch = cents + (size_t)h * C_ * D_;
    for (int i = threadIdx.x; i < C_ * D_; i += blockDim.x) cent[i] = ch[i];
    __syncthreads();
    if (threadIdx.x < C_) {
        float ss = 0.f;
        #pragma unroll
        for (int d = 0; d < D_; ++d) {
            float v = cent[threadIdx.x * D_ + d];
            ss += v * v;
        }
        cinv[threadIdx.x] = 1.0f / fmaxf(sqrtf(ss), EPSN);
    }
    __syncthreads();

    const float inv_t = 1.0f / fabsf(temp[0]);
    const int   s     = blockIdx.x * blockDim.x + threadIdx.x;   // row in [0,S)

    const float* row = x + ((size_t)bh * S_ + s) * (size_t)D_;
    float q[D_];
    #pragma unroll
    for (int d4 = 0; d4 < D_ / 4; ++d4) {
        v4f v = *(const v4f*)(row + d4 * 4);
        q[d4 * 4 + 0] = v[0]; q[d4 * 4 + 1] = v[1];
        q[d4 * 4 + 2] = v[2]; q[d4 * 4 + 3] = v[3];
    }

    float ss = 0.f;
    #pragma unroll
    for (int d = 0; d < D_; ++d) ss += q[d] * q[d];
    const float qinv = 1.0f / fmaxf(sqrtf(ss), EPSN);

    float sim[C_];
    float mx = -__builtin_inff();
    #pragma unroll
    for (int c = 0; c < C_; ++c) {
        float dot = 0.f;
        #pragma unroll
        for (int d = 0; d < D_; ++d) dot += q[d] * cent[c * D_ + d];
        sim[c] = dot * qinv * cinv[c] * inv_t;
        mx = fmaxf(mx, sim[c]);
    }
    float sum = 0.f;
    #pragma unroll
    for (int c = 0; c < C_; ++c) { sim[c] = __expf(sim[c] - mx); sum += sim[c]; }
    const float isum = 1.0f / sum;

    _Float16* dst = assign + ((size_t)bh * S_ + s) * (size_t)C_;
    #pragma unroll
    for (int c8 = 0; c8 < C_ / 8; ++c8) {
        v8h o;
        #pragma unroll
        for (int j = 0; j < 8; ++j) o[j] = (_Float16)(sim[c8 * 8 + j] * isum);
        *(v8h*)(dst + c8 * 8) = o;
    }
}

// ---------------------------------------------------------------------------
// Kernel 2: affinity = q_assign x k_assign^T via v_wmma_f32_16x16x32_f16
// (K=32 == one WMMA per 16x16 tile), fused threshold-mask of attn_scores.
// One wave per 16x16 output tile; 8 waves / 256-thread block.
// ---------------------------------------------------------------------------
__global__ __launch_bounds__(256)
void affinity_mask_kernel(const _Float16* __restrict__ qa,   // [B,H,S,C]
                          const _Float16* __restrict__ ka,   // [B,H,S,C]
                          const float*    __restrict__ attn, // [B,H,S,S]
                          float*          __restrict__ out)  // [B,H,S,S]
{
    const int lane = threadIdx.x & 31;
    const int wid  = threadIdx.x >> 5;
    const int bh   = blockIdx.y;

    const int tiles_n = S_ / 16;                 // 128
    const int tile    = blockIdx.x * 8 + wid;    // [0, 128*128)
    const int tm      = (tile / tiles_n) * 16;
    const int tn      = (tile % tiles_n) * 16;

    const _Float16* qab = qa + (size_t)bh * S_ * C_;
    const _Float16* kab = ka + (size_t)bh * S_ * C_;

    const int half = lane >> 4;                  // 0 | 1
    const int l15  = lane & 15;

    // A matrix 16x32 f16: lane holds row M=lane%16, K = 16*half + 0..15
    // (same contiguous K order as B).
    const _Float16* arow = qab + (size_t)(tm + l15) * C_ + 16 * half;
    const v8h a_lo = *(const v8h*)(arow);        // K = koff + 0..7
    const v8h a_hi = *(const v8h*)(arow + 8);    // K = koff + 8..15
    v16h A;
    #pragma unroll
    for (int i = 0; i < 8; ++i) { A[i] = a_lo[i]; A[i + 8] = a_hi[i]; }

    // B matrix 32x16 f16: lane holds column N=lane%16, K = 16*half + 0..15.
    // B[k][n] = k_assign[n][k] -> a contiguous row of k_assign. No transpose.
    const _Float16* brow = kab + (size_t)(tn + l15) * C_ + 16 * half;
    const v16h Bv = *(const v16h*)(brow);

    v8f acc = {};
    acc = __builtin_amdgcn_wmma_f32_16x16x32_f16(
        /*neg_a=*/false, A, /*neg_b=*/false, Bv,
        /*c_mod=*/(short)0, acc, /*reuse_a=*/false, /*reuse_b=*/false);

    // C/D layout: lane covers col N = l15, rows M = 8*half + r (r = VGPR idx).
    const size_t base = (size_t)bh * S_ * S_;
    const int    col  = tn + l15;
    #pragma unroll
    for (int r = 0; r < 8; ++r) {
        const int    row = tm + 8 * half + r;
        const size_t idx = base + (size_t)row * S_ + col;
        const float  sc  = attn[idx];
        out[idx] = (acc[r] > THR) ? sc : -__builtin_inff();
    }
}

// ---------------------------------------------------------------------------
extern "C" void kernel_launch(void* const* d_in, const int* in_sizes, int n_in,
                              void* d_out, int out_size, void* d_ws, size_t ws_size,
                              hipStream_t stream) {
    const float* queries = (const float*)d_in[0];   // [B,H,S,D]
    const float* keys    = (const float*)d_in[1];   // [B,H,S,D]
    const float* attn    = (const float*)d_in[2];   // [B,H,S,S]
    const float* cents   = (const float*)d_in[3];   // [H,C,D]
    const float* temp    = (const float*)d_in[4];   // [1]
    float*       out     = (float*)d_out;           // [B,H,S,S]

    _Float16* qa = (_Float16*)d_ws;                         // 2 MB
    _Float16* ka = qa + (size_t)B_ * H_ * S_ * C_;          // 2 MB

    dim3 g1(S_ / 256, B_ * H_);
    assign_kernel<<<g1, 256, 0, stream>>>(queries, cents, temp, qa);
    assign_kernel<<<g1, 256, 0, stream>>>(keys,    cents, temp, ka);

    dim3 g2((S_ / 16) * (S_ / 16) / 8, B_ * H_);            // 2048 x 16
    affinity_mask_kernel<<<g2, 256, 0, stream>>>(qa, ka, attn, out);
}
